// LinearAttn_7808250544706
// MI455X (gfx1250) — compile-verified, matrix operand order checked by reading, not run on hardware
//
#include <hip/hip_runtime.h>
#include <hip/hip_bf16.h>

// ---------------------------------------------------------------------------
// LinearAttn for MI455X (gfx1250), wave32 + v_wmma_f32_16x16x32_bf16.
//   xb     = bf16(x)                        (pre-converted once, [B,S,C])
//   logits = x @ W + b            [B,S,H]   (stored transposed [B,H,S], f32)
//   a      = softmax_S(logits)              (stored transposed [B,H,S], bf16)
//   ctx    = a^T @ x              [B,H,C]   (f32 split-K atomics -> bf16)
//   out    = a @ ctx              [B,S,C]   (f32, d_out)
// Workspace use: ~115 MB.
// ---------------------------------------------------------------------------

#define B_ 8
#define S_ 8192
#define C_ 512
#define H_ 128

typedef __bf16 bf16_t;
typedef __attribute__((ext_vector_type(16))) __bf16 v16bf;
typedef __attribute__((ext_vector_type(8)))  float  v8f;
typedef __attribute__((ext_vector_type(4)))  int    v4i_t;

typedef __attribute__((address_space(1))) v4i_t* gv4i_p;
typedef __attribute__((address_space(3))) v4i_t* lv4i_p;

__device__ __forceinline__ bf16_t f2bf(float f) { return (bf16_t)f; }

// --- async global->LDS staging (CDNA5 GLOBAL_LOAD_ASYNC_TO_LDS_B128) -------
#if defined(__has_builtin)
#  if __has_builtin(__builtin_amdgcn_global_load_async_to_lds_b128)
#    define HAVE_ASYNC_LDS 1
#  endif
#endif
#ifndef HAVE_ASYNC_LDS
#  define HAVE_ASYNC_LDS 0
#endif

__device__ __forceinline__ void copy_b128_to_lds(const bf16_t* g, bf16_t* l) {
#if HAVE_ASYNC_LDS
    __builtin_amdgcn_global_load_async_to_lds_b128(
        (gv4i_p)g, (lv4i_p)l, /*offset=*/0, /*cpol=*/0);
#else
    *(uint4*)l = *(const uint4*)g;
#endif
}

__device__ __forceinline__ void staging_wait() {
#if HAVE_ASYNC_LDS
#  if __has_builtin(__builtin_amdgcn_s_wait_asynccnt)
    __builtin_amdgcn_s_wait_asynccnt(0);
#  else
    asm volatile("s_wait_asynccnt 0x0" ::: "memory");
#  endif
#endif
}

// Load one 16x32 bf16 WMMA fragment from an LDS tile laid out row-major
// [16+ rows][ld] ("row" = M for A, N for B; B tiles staged transposed [col][k]).
// CDNA5 16-bit layout: lanes 0-15 row=lane, K {0..7, 16..23};
// lanes 16-31 same rows, K {8..15, 24..31}.  ld % 8 == 0, base 16B-aligned.
__device__ __forceinline__ v16bf load_frag(const bf16_t* tile, int ld) {
    const int lane = threadIdx.x & 31;
    const int r    = lane & 15;
    const int half = lane >> 4;
    const bf16_t* p = tile + r * ld + half * 8;
    union { uint4 u[2]; v16bf f; } t;
    t.u[0] = *(const uint4*)(p);
    t.u[1] = *(const uint4*)(p + 16);
    return t.f;
}

#define WMMA_BF16(A, Bf, Cacc) \
    __builtin_amdgcn_wmma_f32_16x16x32_bf16(false, (A), false, (Bf), (short)0, (Cacc), false, false)

// ---------------------------------------------------------------------------
// xcvt: xb = bf16(x)    (33.5M elements, vectorized)
// ---------------------------------------------------------------------------
__global__ __launch_bounds__(256) void xcvt_kernel(const float* __restrict__ x,
                                                   bf16_t* __restrict__ xb) {
    const size_t n4     = (size_t)B_ * S_ * C_ / 4;
    const size_t stride = (size_t)gridDim.x * 256;
    for (size_t i = (size_t)blockIdx.x * 256 + threadIdx.x; i < n4; i += stride) {
        float4 v = ((const float4*)x)[i];
        bf16_t t[4] = { f2bf(v.x), f2bf(v.y), f2bf(v.z), f2bf(v.w) };
        ((uint2*)xb)[i] = *(uint2*)t;
    }
}

// ---------------------------------------------------------------------------
// prep: Wt[h][c] = bf16(W[c][h]);  ctx = 0
// ---------------------------------------------------------------------------
__global__ __launch_bounds__(256) void prep_kernel(const float* __restrict__ W,
                                                   bf16_t* __restrict__ Wt,
                                                   float* __restrict__ ctx) {
    const int idx    = blockIdx.x * 256 + threadIdx.x;
    const int stride = gridDim.x * 256;
    for (int i = idx; i < B_ * H_ * C_; i += stride) ctx[i] = 0.0f;
    for (int i = idx; i < H_ * C_; i += stride) {
        int h = i >> 9;      // /C_
        int c = i & (C_ - 1);
        Wt[i] = f2bf(W[c * H_ + h]);
    }
}

// ---------------------------------------------------------------------------
// gemm1: logits_t[b][h][s] = sum_c xb[b][s][c] * Wt[h][c] + bias[h]
// grid = (B*S)/128 = 512 blocks, 256 threads (8 waves).
// Block tile: 128 (s) x 128 (h), K = C staged in BK=64 chunks (async copies).
// ---------------------------------------------------------------------------
__global__ __launch_bounds__(256) void gemm1_kernel(const bf16_t* __restrict__ xb,
                                                    const bf16_t* __restrict__ Wt,
                                                    const float* __restrict__ bias,
                                                    float* __restrict__ logits) {
    __shared__ __attribute__((aligned(16))) bf16_t lA[128 * 72]; // [s][k]
    __shared__ __attribute__((aligned(16))) bf16_t lB[128 * 72]; // [h][k]
    const int tid  = threadIdx.x;
    const int wave = tid >> 5;
    const int lane = tid & 31;
    const int m0   = blockIdx.x * 128;          // global row (b*S + s)

    v8f acc[8] = {};

    for (int kc = 0; kc < C_; kc += 64) {
        // A: 128 rows x 64 k of xb -> LDS (b128 async copies, 4 per thread)
        for (int i = tid; i < 128 * 8; i += 256) {
            int row = i >> 3;
            int k8  = (i & 7) << 3;
            copy_b128_to_lds(xb + (size_t)(m0 + row) * C_ + kc + k8,
                             &lA[row * 72 + k8]);
        }
        // B: Wt rows already [h][c] -> direct async copy
        for (int i = tid; i < 128 * 8; i += 256) {
            int h  = i >> 3;
            int k8 = (i & 7) << 3;
            copy_b128_to_lds(Wt + h * C_ + kc + k8, &lB[h * 72 + k8]);
        }
        staging_wait();
        __syncthreads();
#pragma unroll
        for (int ks = 0; ks < 64; ks += 32) {
            v16bf af = load_frag(&lA[(wave * 16) * 72 + ks], 72);
            v16bf bfr[8];
#pragma unroll
            for (int n = 0; n < 8; ++n)
                bfr[n] = load_frag(&lB[(n * 16) * 72 + ks], 72);
#pragma unroll
            for (int n = 0; n < 8; ++n)
                acc[n] = WMMA_BF16(af, bfr[n], acc[n]);
        }
        __syncthreads();
    }

    const int n_local = lane & 15;
    const int m_base  = (lane >> 4) * 8;
    const int b       = m0 >> 13;                       // /S_
    const int s_base  = (m0 & (S_ - 1)) + wave * 16 + m_base;
#pragma unroll
    for (int n = 0; n < 8; ++n) {
        const int h = n * 16 + n_local;
        const float bh = bias[h];
        float* dst = logits + ((size_t)b * H_ + h) * S_ + s_base;
#pragma unroll
        for (int r = 0; r < 8; ++r) dst[r] = acc[n][r] + bh;  // 8 consecutive s
    }
}

// ---------------------------------------------------------------------------
// softmax over S per (b,h): one block per row of logits_t; values stay in
// registers (32 per thread), block-reduce max then sum, write bf16 a_t.
// ---------------------------------------------------------------------------
__global__ __launch_bounds__(256) void softmax_kernel(const float* __restrict__ logits,
                                                      bf16_t* __restrict__ a_t) {
    const int bh  = blockIdx.x;                 // b*H + h
    const int tid = threadIdx.x;
    const float* Lrow = logits + (size_t)bh * S_;
    bf16_t* Arow = a_t + (size_t)bh * S_;

    float v[32];
    float mx = -3.402823466e38f;
#pragma unroll
    for (int i = 0; i < 32; ++i) {
        v[i] = Lrow[tid + (i << 8)];
        mx = fmaxf(mx, v[i]);
    }
    __shared__ float sred[256];
    sred[tid] = mx;
    __syncthreads();
    for (int off = 128; off > 0; off >>= 1) {
        if (tid < off) sred[tid] = fmaxf(sred[tid], sred[tid + off]);
        __syncthreads();
    }
    mx = sred[0];
    __syncthreads();

    float sum = 0.0f;
#pragma unroll
    for (int i = 0; i < 32; ++i) {
        v[i] = __expf(v[i] - mx);
        sum += v[i];
    }
    sred[tid] = sum;
    __syncthreads();
    for (int off = 128; off > 0; off >>= 1) {
        if (tid < off) sred[tid] += sred[tid + off];
        __syncthreads();
    }
    const float rinv = 1.0f / sred[0];
#pragma unroll
    for (int i = 0; i < 32; ++i) Arow[tid + (i << 8)] = f2bf(v[i] * rinv);
}

// ---------------------------------------------------------------------------
// gemm2: ctx[b][h][c] += sum_s a_t[b][h][s] * xb[b][s][c]   (split-K atomics)
// grid = B * 4 (c-tiles) * 8 (k-splits) = 256 blocks.
// Block tile: M=128 (h) x N=128 (c), K-chunk = 1024 s, staged BK=64.
// ---------------------------------------------------------------------------
__global__ __launch_bounds__(256) void gemm2_kernel(const bf16_t* __restrict__ a_t,
                                                    const bf16_t* __restrict__ xb,
                                                    float* __restrict__ ctx) {
    __shared__ __attribute__((aligned(16))) bf16_t lA[128 * 72]; // [h][k=s]
    __shared__ __attribute__((aligned(16))) bf16_t lB[128 * 72]; // [c][k=s]
    const int tid  = threadIdx.x;
    const int wave = tid >> 5;
    const int lane = tid & 31;
    const int b    = blockIdx.x >> 5;
    const int rest = blockIdx.x & 31;
    const int c0   = (rest >> 3) * 128;
    const int sbeg = (rest & 7) * 1024;

    v8f acc[8] = {};

    for (int kc = 0; kc < 1024; kc += 64) {
        const int skc = sbeg + kc;
        // A: a_t already [h][s] -> direct async copy
        for (int i = tid; i < 128 * 8; i += 256) {
            int h  = i >> 3;
            int k8 = (i & 7) << 3;
            copy_b128_to_lds(a_t + ((size_t)b * H_ + h) * S_ + skc + k8,
                             &lA[h * 72 + k8]);
        }
        // L2 prefetch of the next chunk's x rows
        if (kc + 64 < 1024) {
            __builtin_prefetch(xb + ((size_t)b * S_ + skc + 64 + (tid & 63)) * C_ + c0, 0, 1);
        }
        // B: transpose xb rows [s][c] -> lB[c][s]
        for (int i = tid; i < 64 * 64; i += 256) {
            int srow = i >> 6;
            int c2   = (i & 63) << 1;
            const bf16_t* p = xb + ((size_t)b * S_ + skc + srow) * C_ + c0 + c2;
            lB[(c2    ) * 72 + srow] = p[0];
            lB[(c2 + 1) * 72 + srow] = p[1];
        }
        staging_wait();
        __syncthreads();
#pragma unroll
        for (int ks = 0; ks < 64; ks += 32) {
            v16bf af = load_frag(&lA[(wave * 16) * 72 + ks], 72);
            v16bf bfr[8];
#pragma unroll
            for (int n = 0; n < 8; ++n)
                bfr[n] = load_frag(&lB[(n * 16) * 72 + ks], 72);
#pragma unroll
            for (int n = 0; n < 8; ++n)
                acc[n] = WMMA_BF16(af, bfr[n], acc[n]);
        }
        __syncthreads();
    }

    const int n_local = lane & 15;
    const int m_base  = (lane >> 4) * 8;
    const int h_base  = wave * 16 + m_base;
#pragma unroll
    for (int n = 0; n < 8; ++n) {
        const int c = c0 + n * 16 + n_local;
#pragma unroll
        for (int r = 0; r < 8; ++r) {
            unsafeAtomicAdd(&ctx[((size_t)b * H_ + h_base + r) * C_ + c], acc[n][r]);
        }
    }
}

// ---------------------------------------------------------------------------
// ctxcvt: ctx f32 -> bf16 (same [b][h][c] layout)
// ---------------------------------------------------------------------------
__global__ __launch_bounds__(256) void ctxcvt_kernel(const float* __restrict__ ctx,
                                                     bf16_t* __restrict__ ctxb) {
    const int idx    = blockIdx.x * 256 + threadIdx.x;
    const int stride = gridDim.x * 256;
    for (int i = idx; i < B_ * H_ * C_; i += stride) ctxb[i] = f2bf(ctx[i]);
}

// ---------------------------------------------------------------------------
// gemm3: out[b][s][c] = sum_h a_t[b][h][s] * ctxb[b][h][c]
// grid = B * 64 (s-tiles) * 4 (c-tiles) = 2048 blocks; K = H = 128 (BK=64).
// ---------------------------------------------------------------------------
__global__ __launch_bounds__(256) void gemm3_kernel(const bf16_t* __restrict__ a_t,
                                                    const bf16_t* __restrict__ ctxb,
                                                    float* __restrict__ out) {
    __shared__ __attribute__((aligned(16))) bf16_t lA[128 * 72]; // [s][k=h]
    __shared__ __attribute__((aligned(16))) bf16_t lB[128 * 72]; // [c][k=h]
    const int tid  = threadIdx.x;
    const int wave = tid >> 5;
    const int lane = tid & 31;
    const int b    = blockIdx.x >> 8;
    const int rest = blockIdx.x & 255;
    const int s0   = (rest >> 2) * 128;
    const int c0   = (rest & 3) * 128;

    v8f acc[8] = {};

    for (int kc = 0; kc < H_; kc += 64) {
        // A: transpose a_t[h][s] -> lA[s][h]   (with L2 prefetch of next rows)
        if (kc + 64 < H_ && tid < 64) {
            __builtin_prefetch(a_t + ((size_t)b * H_ + kc + 64 + tid) * S_ + s0, 0, 1);
        }
        for (int i = tid; i < 64 * 64; i += 256) {
            int hrow = i >> 6;
            int s2   = (i & 63) << 1;
            const bf16_t* p = a_t + ((size_t)b * H_ + kc + hrow) * S_ + s0 + s2;
            lA[(s2    ) * 72 + hrow] = p[0];
            lA[(s2 + 1) * 72 + hrow] = p[1];
        }
        // B: transpose ctxb[h][c] -> lB[c][h]
        for (int i = tid; i < 64 * 64; i += 256) {
            int hrow = i >> 6;
            int c2   = (i & 63) << 1;
            const bf16_t* p = ctxb + ((size_t)b * H_ + kc + hrow) * C_ + c0 + c2;
            lB[(c2    ) * 72 + hrow] = p[0];
            lB[(c2 + 1) * 72 + hrow] = p[1];
        }
        __syncthreads();
#pragma unroll
        for (int ks = 0; ks < 64; ks += 32) {
            v16bf af = load_frag(&lA[(wave * 16) * 72 + ks], 72);
            v16bf bfr[8];
#pragma unroll
            for (int n = 0; n < 8; ++n)
                bfr[n] = load_frag(&lB[(n * 16) * 72 + ks], 72);
#pragma unroll
            for (int n = 0; n < 8; ++n)
                acc[n] = WMMA_BF16(af, bfr[n], acc[n]);
        }
        __syncthreads();
    }

    const int n_local = lane & 15;
    const int m_base  = (lane >> 4) * 8;
    const int s_row   = s0 + wave * 16 + m_base;
#pragma unroll
    for (int n = 0; n < 8; ++n) {
        const int c = c0 + n * 16 + n_local;
#pragma unroll
        for (int r = 0; r < 8; ++r) {
            out[((size_t)b * S_ + s_row + r) * C_ + c] = acc[n][r];
        }
    }
}

// ---------------------------------------------------------------------------
extern "C" void kernel_launch(void* const* d_in, const int* in_sizes, int n_in,
                              void* d_out, int out_size, void* d_ws, size_t ws_size,
                              hipStream_t stream) {
    const float* x    = (const float*)d_in[0];   // [B,S,C] f32
    const float* W    = (const float*)d_in[1];   // [C,H]   f32
    const float* bias = (const float*)d_in[2];   // [H]     f32
    float* out        = (float*)d_out;           // [B,S,C] f32

    // Workspace carve-up (~115 MB total)
    char* ws = (char*)d_ws;
    const size_t XB_BYTES   = (size_t)B_ * S_ * C_ * sizeof(bf16_t);     // 64 MB
    const size_t WT_BYTES   = (size_t)H_ * C_ * sizeof(bf16_t);          // 128 KB
    const size_t LOG_BYTES  = (size_t)B_ * H_ * S_ * sizeof(float);      // 32 MB
    const size_t AT_BYTES   = (size_t)B_ * H_ * S_ * sizeof(bf16_t);     // 16 MB
    const size_t CTX_BYTES  = (size_t)B_ * H_ * C_ * sizeof(float);      // 2 MB

    bf16_t* xbuf   = (bf16_t*)(ws);
    bf16_t* Wt     = (bf16_t*)(ws + XB_BYTES);
    float*  logits = (float*)(ws + XB_BYTES + WT_BYTES);
    bf16_t* a_t    = (bf16_t*)(ws + XB_BYTES + WT_BYTES + LOG_BYTES);
    float*  ctx    = (float*)(ws + XB_BYTES + WT_BYTES + LOG_BYTES + AT_BYTES);
    bf16_t* ctxb   = (bf16_t*)(ws + XB_BYTES + WT_BYTES + LOG_BYTES + AT_BYTES + CTX_BYTES);

    xcvt_kernel   <<<4096, 256, 0, stream>>>(x, xbuf);
    prep_kernel   <<<256,  256, 0, stream>>>(W, Wt, ctx);
    gemm1_kernel  <<<(B_ * S_) / 128, 256, 0, stream>>>(xbuf, Wt, bias, logits);
    softmax_kernel<<<B_ * H_, 256, 0, stream>>>(logits, a_t);
    gemm2_kernel  <<<B_ * 4 * 8, 256, 0, stream>>>(a_t, xbuf, ctx);
    ctxcvt_kernel <<<256, 256, 0, stream>>>(ctx, ctxb);
    gemm3_kernel  <<<B_ * 64 * 4, 256, 0, stream>>>(a_t, ctxb, out);
}
